// PerformerAttention_13932873908385
// MI455X (gfx1250) — matrix operand order repaired and checked by reading, not compiled, not run on hardware
//
#include <hip/hip_runtime.h>
#include <hip/hip_bf16.h>
#include <math.h>

// ---------------------------------------------------------------------------
// Problem constants (from reference)
// ---------------------------------------------------------------------------
#define BATCH 2
#define SEQ   2048
#define HID   2048
#define NH    16
#define NHKV  8
#define HD    128     // head dim D == F
#define M1    (BATCH*SEQ)          // 4096 tokens
#define ROWS_Q (M1*NH)             // 65536 token-heads (q)
#define ROWS_K (M1*NHKV)           // 32768 token-heads (k)

typedef __attribute__((ext_vector_type(16))) __bf16 v16bf;
typedef __attribute__((ext_vector_type(8)))  float  v8f;
typedef int b128_t __attribute__((vector_size(16)));   // v4i, matches builtin param

// ---------------------------------------------------------------------------
// bf16 helpers (round-to-nearest-even)
// ---------------------------------------------------------------------------
__device__ __forceinline__ unsigned short f2bf(float x) {
    unsigned u  = __builtin_bit_cast(unsigned, x);
    unsigned rb = ((u >> 16) & 1u) + 0x7FFFu;
    return (unsigned short)((u + rb) >> 16);
}
__device__ __forceinline__ float bf2f(unsigned short h) {
    unsigned u = ((unsigned)h) << 16;
    return __builtin_bit_cast(float, u);
}

// ---------------------------------------------------------------------------
// gfx1250 async global->LDS copy (ASYNCcnt path).
// Builtin signature (from hipcc diagnostic): (v4i AS1*, v4i AS3*, Imm, Imm)
// ---------------------------------------------------------------------------
__device__ __forceinline__ void async_copy_b128(unsigned short* dst,
                                                const unsigned short* src) {
#if __has_builtin(__builtin_amdgcn_global_load_async_to_lds_b128)
    unsigned short* s = const_cast<unsigned short*>(src);
    __builtin_amdgcn_global_load_async_to_lds_b128(
        (__attribute__((address_space(1))) b128_t*)s,
        (__attribute__((address_space(3))) b128_t*)dst,
        0, 0);
#else
    asm volatile("global_load_async_to_lds_b128 %0, %1, off"
                 :: "v"((unsigned)(unsigned long long)
                        (__attribute__((address_space(3))) unsigned short*)dst),
                    "v"(src)
                 : "memory");
#endif
}

__device__ __forceinline__ void async_wait0() {
#if __has_builtin(__builtin_amdgcn_s_wait_asynccnt)
    __builtin_amdgcn_s_wait_asynccnt(0);
#else
    asm volatile("s_wait_asynccnt 0x0" ::: "memory");
#endif
}

// ---------------------------------------------------------------------------
// f32 -> bf16 conversion (grid-stride)
// ---------------------------------------------------------------------------
__global__ void cvt_f32_bf16(const float* __restrict__ in,
                             unsigned short* __restrict__ out, long n) {
    long i = (long)blockIdx.x * blockDim.x + threadIdx.x;
    long stride = (long)gridDim.x * blockDim.x;
    for (; i < n; i += stride) out[i] = f2bf(in[i]);
}

// ---------------------------------------------------------------------------
// Generic bf16 WMMA GEMM: C[M,N] = A[M,K] * B[N,K]^T (+ bias[N])
// Both A and B row-major, K contiguous (the "x @ W.T" form).
// 256 threads = 8 waves (wave32); K step 32; double-buffered LDS filled by
// async global->LDS b128 copies.
//   wave tile = (16*WMT) x (16*WNT);  WAVES_M*WAVES_N must equal 8.
// ---------------------------------------------------------------------------
#define BK 32

template <int TBM, int TBN, int WMT, int WNT>
__global__ __launch_bounds__(256, 2)
void gemm_bf16_wmma(const unsigned short* __restrict__ A,
                    const unsigned short* __restrict__ B,
                    const float* __restrict__ bias,
                    float* __restrict__ C,
                    int M, int N, int K) {
    constexpr int WAVES_M = TBM / (16 * WMT);
    constexpr int WAVES_N = TBN / (16 * WNT);
    static_assert(WAVES_M * WAVES_N == 8, "8 waves per block");
    constexpr int ACH = TBM * 4;            // 16B chunks per A tile
    constexpr int CH  = (TBM + TBN) * 4;    // total chunks per tile pair
    constexpr int NCH = CH / 256;           // chunks per thread (exact)

    __shared__ __align__(16) unsigned short As[2][TBM * BK];
    __shared__ __align__(16) unsigned short Bs[2][TBN * BK];

    const int tid  = threadIdx.x;
    const int lane = tid & 31;
    const int wave = tid >> 5;
    const int half = lane >> 4;      // 0: lanes 0-15, 1: lanes 16-31
    const int l16  = lane & 15;

    const int wm = (wave % WAVES_M) * (16 * WMT);
    const int wn = (wave / WAVES_M) * (16 * WNT);

    const int m0 = blockIdx.x * TBM;
    const int n0 = blockIdx.y * TBN;

    // Async fill of one double-buffer slot: each thread moves NCH b128 chunks.
    auto loadTile = [&](int buf, int k0) {
#pragma unroll
        for (int i = 0; i < NCH; ++i) {
            const int c = tid + i * 256;
            unsigned short* dst;
            const unsigned short* src;
            if (c < ACH) {
                const int r = c >> 2, cc = (c & 3) * 8;
                dst = &As[buf][r * BK + cc];
                src = A + (size_t)(m0 + r) * K + k0 + cc;
            } else {
                const int c2 = c - ACH;
                const int r = c2 >> 2, cc = (c2 & 3) * 8;
                dst = &Bs[buf][r * BK + cc];
                src = B + (size_t)(n0 + r) * K + k0 + cc;
            }
            async_copy_b128(dst, src);
        }
    };

    v8f acc[WMT][WNT];
#pragma unroll
    for (int a = 0; a < WMT; ++a)
#pragma unroll
        for (int b = 0; b < WNT; ++b) acc[a][b] = (v8f){};

    const int nk = K / BK;
    loadTile(0, 0);
    async_wait0();
    __syncthreads();

    union Frag { uint4 u[2]; v16bf v; };

    for (int kt = 0; kt < nk; ++kt) {
        const int cur = kt & 1;
        if (kt + 1 < nk) loadTile(cur ^ 1, (kt + 1) * BK);
        if (kt + 2 < nk) {          // gfx1250 global_prefetch_b8, tile after next
            __builtin_prefetch(A + (size_t)(m0 + (tid >> 2)) * K + (kt + 2) * BK, 0, 3);
            __builtin_prefetch(B + (size_t)(n0 + (tid & 63)) * K + (kt + 2) * BK, 0, 3);
        }

        // A fragments (16x32 bf16, ISA 7.12.2):
        //  half==0: K = {0..7} U {16..23}; half==1: K = {8..15} U {24..31}
        Frag af[WMT];
#pragma unroll
        for (int a = 0; a < WMT; ++a) {
            const unsigned short* arow = &As[cur][(wm + a * 16 + l16) * BK];
            af[a].u[0] = *(const uint4*)(arow + half * 8);
            af[a].u[1] = *(const uint4*)(arow + 16 + half * 8);
        }
        // B fragments (32x16): lane holds column n=l16, K = {half*8 + 0..7} U {16 + half*8 + 0..7}
        Frag bfr[WNT];
#pragma unroll
        for (int b = 0; b < WNT; ++b) {
            const unsigned short* brow = &Bs[cur][(wn + b * 16 + l16) * BK + half * 8];
            bfr[b].u[0] = *(const uint4*)(brow);
            bfr[b].u[1] = *(const uint4*)(brow + 16);
        }
#pragma unroll
        for (int a = 0; a < WMT; ++a)
#pragma unroll
            for (int b = 0; b < WNT; ++b)
                acc[a][b] = __builtin_amdgcn_wmma_f32_16x16x32_bf16(
                    false, af[a].v, false, bfr[b].v, (short)0, acc[a][b], false, false);

        async_wait0();
        __syncthreads();
    }

    // Epilogue. C/D layout: VGPR r -> M = half*8 + r, N = l16.
#pragma unroll
    for (int b = 0; b < WNT; ++b) {
        const int cn = n0 + wn + b * 16 + l16;
        const float bb = bias ? bias[cn] : 0.0f;
#pragma unroll
        for (int a = 0; a < WMT; ++a) {
            const int mb = m0 + wm + a * 16 + half * 8;
#pragma unroll
            for (int r = 0; r < 8; ++r)
                C[(size_t)(mb + r) * N + cn] = acc[a][b][r] + bb;
        }
    }
}

// ---------------------------------------------------------------------------
// RoPE + per-(token,head) L2 norm. One block (128 threads) per token-head.
// in : f32 [token][head][128]   (GEMM output layout)
// out: bf16 [token*nheads + head][128]   norms: f32 per row
// ---------------------------------------------------------------------------
__global__ __launch_bounds__(128)
void rope_norm_kernel(const float* __restrict__ in,
                      unsigned short* __restrict__ out,
                      float* __restrict__ norms, int nheads) {
    const int blk = blockIdx.x;
    const int d   = threadIdx.x;         // 0..127
    const int token = blk / nheads;
    const int h     = blk - token * nheads;
    const int s     = token & (SEQ - 1); // position within sequence

    const float* row = in + (size_t)token * nheads * HD + (size_t)h * HD;

    const int   i    = d & 63;
    const float freq = __powf(10000.0f, -((float)(2 * i)) / 128.0f);
    const float ang  = (float)s * freq;
    float sn, cs;
    __sincosf(ang, &sn, &cs);

    const float x     = row[d];
    const float other = (d < 64) ? -row[d + 64] : row[d - 64];
    const float r     = x * cs + other * sn;

    __shared__ float red[128];
    red[d] = r * r;
    __syncthreads();
#pragma unroll
    for (int off = 64; off > 0; off >>= 1) {
        if (d < off) red[d] += red[d + off];
        __syncthreads();
    }

    out[(size_t)blk * HD + d] = f2bf(r);
    if (d == 0) norms[blk] = sqrtf(red[0]);
}

// ---------------------------------------------------------------------------
// Performer feature epilogue: feats = [sin(p/n), cos(p/n)] / sqrt(F)
// p: f32 [row][64], norms: f32 [row] -> out bf16 [row][128]
// ---------------------------------------------------------------------------
__global__ __launch_bounds__(64)
void feat_kernel(const float* __restrict__ p,
                 const float* __restrict__ norms,
                 unsigned short* __restrict__ out) {
    const int row = blockIdx.x;
    const int e   = threadIdx.x;   // 0..63
    const float n = fmaxf(norms[row], 1e-6f);
    const float v = p[(size_t)row * 64 + e] / n;
    float sv, cv;
    __sincosf(v, &sv, &cv);
    const float sc = 0.08838834764831845f;  // 1/sqrt(128)
    out[(size_t)row * HD + e]      = f2bf(sv * sc);
    out[(size_t)row * HD + 64 + e] = f2bf(cv * sc);
}

// ---------------------------------------------------------------------------
// Per-position combine. The reference einsums contract over heads per
// position:  out[h,d] = 2 * sum_hk ( qp[h]·kp[hk] ) * v[hk,d]
// One block (256 threads) per position.
// ---------------------------------------------------------------------------
__global__ __launch_bounds__(256)
void attn_kernel(const unsigned short* __restrict__ qp,
                 const unsigned short* __restrict__ kp,
                 const float* __restrict__ v,
                 unsigned short* __restrict__ out) {
    const int pos = blockIdx.x;
    const int t   = threadIdx.x;

    __shared__ float qs[NH * HD];    // 16x128
    __shared__ float ks[NHKV * HD];  //  8x128
    __shared__ float Ssc[NH * NHKV]; // 16x8

    const unsigned short* qb = qp + (size_t)pos * NH * HD;
    const unsigned short* kb = kp + (size_t)pos * NHKV * HD;
    for (int i = t; i < NH * HD; i += 256)   qs[i] = bf2f(qb[i]);
    for (int i = t; i < NHKV * HD; i += 256) ks[i] = bf2f(kb[i]);
    __syncthreads();

    if (t < NH * NHKV) {
        const int h = t >> 3, hk = t & 7;
        float acc = 0.0f;
#pragma unroll 4
        for (int f = 0; f < HD; ++f) acc += qs[h * HD + f] * ks[hk * HD + f];
        Ssc[t] = acc;
    }
    __syncthreads();

    const int h  = t >> 4;
    const int d0 = (t & 15) * 8;
    const float* vb = v + (size_t)pos * NHKV * HD;
    float o[8] = {0, 0, 0, 0, 0, 0, 0, 0};
#pragma unroll
    for (int hk = 0; hk < NHKV; ++hk) {
        const float sc = Ssc[h * NHKV + hk];
#pragma unroll
        for (int j = 0; j < 8; ++j) o[j] += sc * vb[hk * HD + d0 + j];
    }
#pragma unroll
    for (int j = 0; j < 8; ++j)
        out[(size_t)pos * HID + h * HD + d0 + j] = f2bf(2.0f * o[j]);
}

// ---------------------------------------------------------------------------
// Host orchestration
// ---------------------------------------------------------------------------
static inline size_t align256(size_t x) { return (x + 255) & ~(size_t)255; }

extern "C" void kernel_launch(void* const* d_in, const int* in_sizes, int n_in,
                              void* d_out, int out_size, void* d_ws, size_t ws_size,
                              hipStream_t stream) {
    (void)in_sizes; (void)n_in; (void)out_size; (void)ws_size;

    const float* hs = (const float*)d_in[0];
    const float* Wq = (const float*)d_in[1];
    const float* bq = (const float*)d_in[2];
    const float* Wk = (const float*)d_in[3];
    const float* bk = (const float*)d_in[4];
    const float* Wv = (const float*)d_in[5];
    const float* bv = (const float*)d_in[6];
    const float* Wo = (const float*)d_in[7];
    const float* bo = (const float*)d_in[8];
    const float* Rm = (const float*)d_in[9];
    float* outp = (float*)d_out;

    char* ws = (char*)d_ws;
    size_t off = 0;
    auto take = [&](size_t bytes) { size_t r = off; off += align256(bytes); return r; };

    unsigned short* hsb = (unsigned short*)(ws + take((size_t)M1 * HID * 2));
    unsigned short* wqb = (unsigned short*)(ws + take((size_t)HID * HID * 2));
    unsigned short* wkb = (unsigned short*)(ws + take((size_t)NHKV * HD * HID * 2));
    unsigned short* wvb = (unsigned short*)(ws + take((size_t)NHKV * HD * HID * 2));
    unsigned short* wob = (unsigned short*)(ws + take((size_t)HID * HID * 2));
    unsigned short* rb  = (unsigned short*)(ws + take((size_t)64 * HD * 2));
    float* qf = (float*)(ws + take((size_t)M1 * HID * 4));        // reused for pq
    float* kf = (float*)(ws + take((size_t)M1 * NHKV * HD * 4));  // reused for pk
    float* vf = (float*)(ws + take((size_t)M1 * NHKV * HD * 4));
    unsigned short* qr = (unsigned short*)(ws + take((size_t)ROWS_Q * HD * 2));
    unsigned short* kr = (unsigned short*)(ws + take((size_t)ROWS_K * HD * 2));
    float* nq = (float*)(ws + take((size_t)ROWS_Q * 4));
    float* nk = (float*)(ws + take((size_t)ROWS_K * 4));
    unsigned short* qpb = (unsigned short*)(ws + take((size_t)ROWS_Q * HD * 2));
    unsigned short* kpb = (unsigned short*)(ws + take((size_t)ROWS_K * HD * 2));
    unsigned short* aob = (unsigned short*)(ws + take((size_t)M1 * HID * 2));

    // 1) f32 -> bf16 packing of activations + weights
    cvt_f32_bf16<<<4096, 256, 0, stream>>>(hs, hsb, (long)M1 * HID);
    cvt_f32_bf16<<<4096, 256, 0, stream>>>(Wq, wqb, (long)HID * HID);
    cvt_f32_bf16<<<2048, 256, 0, stream>>>(Wk, wkb, (long)NHKV * HD * HID);
    cvt_f32_bf16<<<2048, 256, 0, stream>>>(Wv, wvb, (long)NHKV * HD * HID);
    cvt_f32_bf16<<<4096, 256, 0, stream>>>(Wo, wob, (long)HID * HID);
    cvt_f32_bf16<<<32,   256, 0, stream>>>(Rm, rb,  (long)64 * HD);

    // 2) Q/K/V projections (bf16 WMMA, f32 accum); big tile: wave = 32x32
    gemm_bf16_wmma<64, 128, 2, 2>
        <<<dim3(M1 / 64, HID / 128), 256, 0, stream>>>(hsb, wqb, bq, qf, M1, HID, HID);
    gemm_bf16_wmma<64, 128, 2, 2>
        <<<dim3(M1 / 64, (NHKV * HD) / 128), 256, 0, stream>>>(hsb, wkb, bk, kf, M1, NHKV * HD, HID);
    gemm_bf16_wmma<64, 128, 2, 2>
        <<<dim3(M1 / 64, (NHKV * HD) / 128), 256, 0, stream>>>(hsb, wvb, bv, vf, M1, NHKV * HD, HID);

    // 3) RoPE + norms (norm deferred into the feature epilogue)
    rope_norm_kernel<<<ROWS_Q, 128, 0, stream>>>(qf, qr, nq, NH);
    rope_norm_kernel<<<ROWS_K, 128, 0, stream>>>(kf, kr, nk, NHKV);

    // 4) Random-feature projection p = x_rot @ R^T (N=64 -> small-tile config)
    float* pq = qf;
    float* pk = kf;
    gemm_bf16_wmma<64, 64, 1, 2>
        <<<dim3(ROWS_Q / 64, 1), 256, 0, stream>>>(qr, rb, nullptr, pq, ROWS_Q, 64, HD);
    gemm_bf16_wmma<64, 64, 1, 2>
        <<<dim3(ROWS_K / 64, 1), 256, 0, stream>>>(kr, rb, nullptr, pk, ROWS_K, 64, HD);

    // 5) feats = [sin(p/n), cos(p/n)] / sqrt(F)
    feat_kernel<<<ROWS_Q, 64, 0, stream>>>(pq, nq, qpb);
    feat_kernel<<<ROWS_K, 64, 0, stream>>>(pk, nk, kpb);

    // 6) per-position combine (head contraction)
    attn_kernel<<<M1, 256, 0, stream>>>(qpb, kpb, vf, aob);

    // 7) output projection -> f32 d_out
    gemm_bf16_wmma<64, 128, 2, 2>
        <<<dim3(M1 / 64, HID / 128), 256, 0, stream>>>(aob, wob, bo, outp, M1, HID, HID);
}
